// _EdgeBlock_26474178413323
// MI455X (gfx1250) — matrix operand order repaired and checked by reading, not compiled, not run on hardware
//
#include <hip/hip_runtime.h>

// ---------------------------------------------------------------------------
// CDNA5 (gfx1250) edge-MLP: bf16 WMMA (v_wmma_f32_16x16x32_bf16), wave32.
// ---------------------------------------------------------------------------

typedef __attribute__((ext_vector_type(16))) __bf16 v16bf;
typedef __attribute__((ext_vector_type(8)))  float  v8f;

union Frag16 {
    v16bf v;
    uint4 q[2];
};

__device__ __forceinline__ unsigned short f2bf(float f) {
    unsigned int x = __float_as_uint(f);
    unsigned int r = (x + 0x7fffu + ((x >> 16) & 1u)) >> 16;  // round-nearest-even
    return (unsigned short)r;
}

// ---------------------------------------------------------------------------
// Pack fp32 weights [K x 128] into bf16 WMMA B-fragment layout:
//   Wp[((kc*8 + nt)*32 + lane)*16 + h] = bf16( W[kc*32 + (lane>>4)*16 + h][nt*16 + (lane&15)] )
// so each lane's per-kc B fragment is 32 contiguous bytes.
// ---------------------------------------------------------------------------
__global__ void pack_w_kernel(const float* __restrict__ W,
                              unsigned short* __restrict__ Wp, int KC) {
    int idx = blockIdx.x * 256 + threadIdx.x;      // one thread per (kc, nt, lane)
    int total = KC * 8 * 32;
    if (idx >= total) return;
    int lane = idx & 31;
    int nt   = (idx >> 5) & 7;
    int kc   = idx >> 8;
    int n     = nt * 16 + (lane & 15);
    int kbase = kc * 32 + (lane >> 4) * 16;
    unsigned short* dst = Wp + (size_t)idx * 16;
#pragma unroll
    for (int h = 0; h < 16; ++h)
        dst[h] = f2bf(W[(size_t)(kbase + h) * 128 + n]);
}

// ---------------------------------------------------------------------------
// One MLP layer worth of WMMAs for one wave / one 16-edge M-tile.
// A fragment (16x32 bf16): lane holds M = lane&15,
//   halves 0..15 -> K = kc*32 + (lane>>4)*16 + 0..15 (matches packed B)
// K-contiguous -> two b128 LDS loads.
// ---------------------------------------------------------------------------
template <int KC>
__device__ __forceinline__ v8f mlp_layer(const unsigned short* act_lds, int act_stride,
                                         const unsigned short* __restrict__ Wp,
                                         int nt, int lane, int mt) {
    const int lmod  = lane & 15;
    const int lhalf = lane >> 4;
    const int m     = mt * 16 + lmod;
    v8f acc = {};
#pragma unroll
    for (int kc = 0; kc < KC; ++kc) {
        Frag16 a, b;
        const int kb = kc * 32 + lhalf * 16;
        a.q[0] = *(const uint4*)&act_lds[m * act_stride + kb];
        a.q[1] = *(const uint4*)&act_lds[m * act_stride + kb + 8];
        const uint4* wp = (const uint4*)(Wp + (((size_t)(kc * 8 + nt)) * 32 + lane) * 16);
        b.q[0] = wp[0];
        b.q[1] = wp[1];
        acc = __builtin_amdgcn_wmma_f32_16x16x32_bf16(
            false, a.v, false, b.v, (short)0, acc, false, false);
    }
    return acc;
}

// ---------------------------------------------------------------------------
// Main kernel: 256 threads (8 wave32), 32 edges per block.
// Wave w owns output columns [w*16, w*16+16).
// ---------------------------------------------------------------------------
__global__ __launch_bounds__(256)
void edge_mlp_kernel(const float* __restrict__ nodef,
                     const float* __restrict__ edgef,
                     const float* __restrict__ b0,
                     const float* __restrict__ b1,
                     const float* __restrict__ b2,
                     const float* __restrict__ gamma,
                     const float* __restrict__ beta,
                     const int*   __restrict__ src,
                     const int*   __restrict__ dst,
                     const unsigned short* __restrict__ W0p,
                     const unsigned short* __restrict__ W1p,
                     const unsigned short* __restrict__ W2p,
                     float* __restrict__ out, int nE) {
    __shared__ __attribute__((aligned(16))) unsigned short cat_lds[32 * 384];  // 24.0 KB
    __shared__ __attribute__((aligned(16))) unsigned short hA[32 * 128];       //  8.0 KB
    __shared__ __attribute__((aligned(16))) unsigned short hB[32 * 128];       //  8.0 KB
    __shared__ __attribute__((aligned(16))) float          h2[32 * 128];       // 16.0 KB
    __shared__ float s_sum[32], s_ssq[32];

    const int tid  = threadIdx.x;
    const int lane = tid & 31;
    const int wave = tid >> 5;           // 8 waves -> nt
    const int e0   = blockIdx.x * 32;

    if (tid < 32) { s_sum[tid] = 0.f; s_ssq[tid] = 0.f; }

    // ---- Stage 1: gather src/dst/edge features -> bf16 concat tile in LDS ----
    // 32 rows x 3 segments x 32 float4 = 3072 float4; 12 per thread.
#pragma unroll
    for (int i = 0; i < 12; ++i) {
        int flat = tid + i * 256;
        int f4   = flat & 31;
        int rs   = flat >> 5;            // 0..95
        int m    = rs / 3;
        int seg  = rs - m * 3;
        int e    = e0 + m;
        if (e >= nE) e = nE - 1;
        const float* p;
        if (seg == 0)      p = nodef + (size_t)src[e] * 128;
        else if (seg == 1) p = nodef + (size_t)dst[e] * 128;
        else               p = edgef + (size_t)e * 128;
        float4 v = ((const float4*)p)[f4];
        unsigned int lo = (unsigned int)f2bf(v.x) | ((unsigned int)f2bf(v.y) << 16);
        unsigned int hi = (unsigned int)f2bf(v.z) | ((unsigned int)f2bf(v.w) << 16);
        uint2 pk; pk.x = lo; pk.y = hi;
        *(uint2*)&cat_lds[m * 384 + seg * 128 + f4 * 4] = pk;
    }
    __syncthreads();

    const int nt    = wave;
    const int lmod  = lane & 15;
    const int lhalf = lane >> 4;
    const int ncol  = nt * 16 + lmod;

    // ---- Layer 0: [32x384] @ [384x128] + b0, ReLU ----
    {
        const float bias = b0[ncol];
#pragma unroll
        for (int mt = 0; mt < 2; ++mt) {
            v8f acc = mlp_layer<12>(cat_lds, 384, W0p, nt, lane, mt);
            const int mrow = mt * 16 + lhalf * 8;
#pragma unroll
            for (int r = 0; r < 8; ++r) {
                float v = acc[r] + bias;
                v = v > 0.f ? v : 0.f;
                hA[(mrow + r) * 128 + ncol] = f2bf(v);
            }
        }
    }
    __syncthreads();

    // ---- Layer 1: [32x128] @ [128x128] + b1, ReLU ----
    {
        const float bias = b1[ncol];
#pragma unroll
        for (int mt = 0; mt < 2; ++mt) {
            v8f acc = mlp_layer<4>(hA, 128, W1p, nt, lane, mt);
            const int mrow = mt * 16 + lhalf * 8;
#pragma unroll
            for (int r = 0; r < 8; ++r) {
                float v = acc[r] + bias;
                v = v > 0.f ? v : 0.f;
                hB[(mrow + r) * 128 + ncol] = f2bf(v);
            }
        }
    }
    __syncthreads();

    // ---- Layer 2: [32x128] @ [128x128] + b2 (f32 result to LDS) ----
    {
        const float bias = b2[ncol];
#pragma unroll
        for (int mt = 0; mt < 2; ++mt) {
            v8f acc = mlp_layer<4>(hB, 128, W2p, nt, lane, mt);
            const int mrow = mt * 16 + lhalf * 8;
#pragma unroll
            for (int r = 0; r < 8; ++r)
                h2[(mrow + r) * 128 + ncol] = acc[r] + bias;
        }
    }
    __syncthreads();

    // ---- LayerNorm partial sums: 8 threads per edge row ----
    {
        int m = tid >> 3;                // 0..31
        int c = tid & 7;                 // 16-feature chunk
        const float* row = &h2[m * 128 + c * 16];
        float s = 0.f, ss = 0.f;
#pragma unroll
        for (int j = 0; j < 16; ++j) { float v = row[j]; s += v; ss += v * v; }
        atomicAdd(&s_sum[m], s);
        atomicAdd(&s_ssq[m], ss);
    }
    __syncthreads();

    // ---- Normalize + affine + residual, float4 stores ----
    {
        int m = tid >> 3;
        int c = tid & 7;
        int e = e0 + m;
        if (e < nE) {
            float mu  = s_sum[m] * (1.f / 128.f);
            float var = s_ssq[m] * (1.f / 128.f) - mu * mu;
            float rs  = rsqrtf(var + 1e-5f);
            const float* row  = &h2[m * 128 + c * 16];
            const float* erow = edgef + (size_t)e * 128;
            float*       orow = out   + (size_t)e * 128;
#pragma unroll
            for (int qq = 0; qq < 4; ++qq) {
                int n = c * 16 + qq * 4;
                float4 o;
                o.x = (row[qq * 4 + 0] - mu) * rs * gamma[n + 0] + beta[n + 0] + erow[n + 0];
                o.y = (row[qq * 4 + 1] - mu) * rs * gamma[n + 1] + beta[n + 1] + erow[n + 1];
                o.z = (row[qq * 4 + 2] - mu) * rs * gamma[n + 2] + beta[n + 2] + erow[n + 2];
                o.w = (row[qq * 4 + 3] - mu) * rs * gamma[n + 3] + beta[n + 3] + erow[n + 3];
                *(float4*)(orow + n) = o;
            }
        }
    }
}

// ---------------------------------------------------------------------------
extern "C" void kernel_launch(void* const* d_in, const int* in_sizes, int n_in,
                              void* d_out, int out_size, void* d_ws, size_t ws_size,
                              hipStream_t stream) {
    const float* nodef = (const float*)d_in[0];
    const float* edgef = (const float*)d_in[1];
    const float* W0    = (const float*)d_in[2];
    const float* b0    = (const float*)d_in[3];
    const float* W1    = (const float*)d_in[4];
    const float* b1    = (const float*)d_in[5];
    const float* W2    = (const float*)d_in[6];
    const float* b2    = (const float*)d_in[7];
    const float* gamma = (const float*)d_in[8];
    const float* beta  = (const float*)d_in[9];
    const int*   src   = (const int*)d_in[10];
    const int*   dst   = (const int*)d_in[11];
    float*       out   = (float*)d_out;
    const int    nE    = in_sizes[10];   // E = len(src_idx)

    // Workspace: packed bf16 weights (W0: 49152, W1/W2: 16384 halves each).
    unsigned short* W0p = (unsigned short*)d_ws;
    unsigned short* W1p = W0p + 12 * 8 * 32 * 16;
    unsigned short* W2p = W1p + 4 * 8 * 32 * 16;

    pack_w_kernel<<<(12 * 8 * 32 + 255) / 256, 256, 0, stream>>>(W0, W0p, 12);
    pack_w_kernel<<<( 4 * 8 * 32 + 255) / 256, 256, 0, stream>>>(W1, W1p, 4);
    pack_w_kernel<<<( 4 * 8 * 32 + 255) / 256, 256, 0, stream>>>(W2, W2p, 4);

    int nblocks = (nE + 31) / 32;
    edge_mlp_kernel<<<nblocks, 256, 0, stream>>>(nodef, edgef, b0, b1, b2,
                                                 gamma, beta, src, dst,
                                                 W0p, W1p, W2p, out, nE);
}
